// SelfAttention_42700564857190
// MI455X (gfx1250) — matrix operand order repaired and checked by reading, not compiled, not run on hardware
//
#include <hip/hip_runtime.h>

// ---------------------------------------------------------------------------
// CDNA5 (gfx1250) causal self-attention.
//   prep:    x -> bf16 copy;  Wq/Wk/Wv -> bf16, transposed to Wt[n][k]
//   kernel1: QKV GEMM, tiles staged by Tensor Data Mover (tensor_load_to_lds,
//            double buffered, s_wait_tensorcnt), MACs via v_wmma bf16.
//   kernel2: flash attention, split-D across 8 waves, 64-key steps, all WMMA.
// ---------------------------------------------------------------------------

typedef __attribute__((ext_vector_type(16))) __bf16        v16bf;
typedef __attribute__((ext_vector_type(8)))  float         v8f;
typedef __attribute__((ext_vector_type(4)))  unsigned int  u32x4;
typedef __attribute__((ext_vector_type(2)))  unsigned int  u32x2;
typedef __attribute__((ext_vector_type(4)))  float         f32x4;
typedef __attribute__((ext_vector_type(8)))  int           i32x8;
typedef __attribute__((ext_vector_type(4)))  int           i32x4;

#define DMODEL 1024
#define SEQ    2048
#define NBATCH 4
#define MTOT   (NBATCH * SEQ)  // 8192
#define LD1    40              // padded row length (elements) of staged tiles
#define SLD    68              // padded Ssc row stride (floats): 68 % 64 banks = 4
#define PLD    72              // padded Pm row stride (bf16): 36 dwords, %8==0

__device__ __forceinline__ unsigned short f2bf(float f) {
  unsigned int u = __float_as_uint(f);
  u += 0x7FFFu + ((u >> 16) & 1u);  // round-to-nearest-even
  return (unsigned short)(u >> 16);
}

union V16BF { v16bf v; u32x4 q[2]; };

// A fragment (16x32 bf16 MxK), row-major source, row stride ld (elems, %8==0).
__device__ __forceinline__ v16bf frag_a(const unsigned short* base, int lane, int ld) {
  const unsigned short* p = base + (lane & 15) * ld + ((lane & 16) ? 8 : 0);
  V16BF t;
  t.q[0] = *(const u32x4*)(p);
  t.q[1] = *(const u32x4*)(p + 16);
  return t.v;
}

// B fragment (32x16 bf16 KxN), source N-major: row n holds contiguous K.
__device__ __forceinline__ v16bf frag_b(const unsigned short* base, int lane, int ld) {
  const unsigned short* p = base + (lane & 15) * ld + ((lane & 16) ? 8 : 0);
  V16BF t;
  t.q[0] = *(const u32x4*)(p);
  t.q[1] = *(const u32x4*)(p + 16);
  return t.v;
}

__device__ __forceinline__ v8f wmma_bf16(v16bf a, v16bf b, v8f c) {
  return __builtin_amdgcn_wmma_f32_16x16x32_bf16(false, a, false, b, (short)0, c, false, false);
}

__device__ __forceinline__ unsigned lds_off(const void* p) {
  // generic pointer to LDS: low 32 bits are the in-workgroup LDS byte offset
  return (unsigned)(uintptr_t)p;
}

// ---------------------------------------------------------------------------
// Tensor Data Mover: load a 2D tile (tile_w x tile_h, 2-byte elems) from a
// row-major tensor into LDS, padding 4 DWORDs after every 16 DWORDs
// (one 32-elem row) => LDS row stride LD1 = 40 elems. ISA 08 D# groups 0/1.
// ---------------------------------------------------------------------------
__device__ __forceinline__ void tdm_load_tile_2d(
    unsigned lds_byte_addr, const unsigned short* gtile,
    unsigned tensor_w, unsigned tensor_h, unsigned stride_elems,
    unsigned tile_w, unsigned tile_h) {
  unsigned long long ga = (unsigned long long)(uintptr_t)gtile;
  u32x4 g0;
  g0[0] = 1u;                                                 // count=1 valid D#
  g0[1] = lds_byte_addr;                                      // lds_addr
  g0[2] = (unsigned)(ga & 0xFFFFFFFFu);                       // global_addr[31:0]
  g0[3] = (unsigned)((ga >> 32) & 0x1FFFFFFu) | (2u << 30);   // ga[56:32] | type=2
  i32x8 g1;
  g1[0] = (int)((1u << 16)      // data_size = 1 -> 2 bytes
              | (1u << 20)      // pad_enable
              | (3u << 22)      // pad_interval: 16 DWORDs
              | (3u << 25));    // pad_amount:   4 DWORDs
  g1[1] = (int)((tensor_w & 0xFFFFu) << 16);                  // tensor_dim0 lo
  g1[2] = (int)((tensor_w >> 16) | ((tensor_h & 0xFFFFu) << 16));
  g1[3] = (int)((tensor_h >> 16) | (tile_w << 16));           // tile_dim0
  g1[4] = (int)(tile_h & 0xFFFFu);                            // tile_dim1
  g1[5] = (int)stride_elems;                                  // dim0 stride lo
  g1[6] = 0;
  g1[7] = 0;
  i32x4 gz = {0, 0, 0, 0};
#if defined(__clang_major__) && (__clang_major__ >= 23)
  i32x8 gz8 = {0, 0, 0, 0, 0, 0, 0, 0};
  __builtin_amdgcn_tensor_load_to_lds(g0, g1, gz, gz, gz8, 0);
#else
  __builtin_amdgcn_tensor_load_to_lds(g0, g1, gz, gz, 0);
#endif
}

// ---------------------------------------------------------------------------
// Prep kernel A: x fp32 -> bf16, straight copy. 8 elems / thread.
// ---------------------------------------------------------------------------
__global__ __launch_bounds__(256) void convert_x_kernel(
    const float* __restrict__ x, unsigned short* __restrict__ xb) {
  size_t i = ((size_t)blockIdx.x * 256 + threadIdx.x) * 8;
  f32x4 a = *(const f32x4*)(x + i);
  f32x4 b = *(const f32x4*)(x + i + 4);
  unsigned short h[8] = { f2bf(a.x), f2bf(a.y), f2bf(a.z), f2bf(a.w),
                          f2bf(b.x), f2bf(b.y), f2bf(b.z), f2bf(b.w) };
  *(u32x4*)(xb + i) = *(const u32x4*)h;
}

// ---------------------------------------------------------------------------
// Prep kernel B: W[k][n] fp32 -> Wt[z][n][k] bf16 (transpose via LDS tile).
// ---------------------------------------------------------------------------
__global__ __launch_bounds__(256) void convert_w_kernel(
    const float* __restrict__ Wq, const float* __restrict__ Wk,
    const float* __restrict__ Wv, unsigned short* __restrict__ Wt) {
  __shared__ unsigned short t[32 * 40];
  const int z = blockIdx.z;
  const float* W = (z == 0) ? Wq : (z == 1) ? Wk : Wv;
  unsigned short* o = Wt + (size_t)z * DMODEL * DMODEL;
  const int k0 = blockIdx.x * 32, n0 = blockIdx.y * 32;
  const int tid = threadIdx.x;
#pragma unroll
  for (int i = 0; i < 4; i++) {
    int e = tid + 256 * i;
    int r = e >> 5, c = e & 31;     // r = k row, c = n col (coalesced read)
    t[c * 40 + r] = f2bf(W[(size_t)(k0 + r) * DMODEL + n0 + c]);
  }
  __syncthreads();
#pragma unroll
  for (int i = 0; i < 4; i++) {
    int e = tid + 256 * i;
    int r = e >> 5, c = e & 31;     // r = n row, c = k col (coalesced write)
    o[(size_t)(n0 + r) * DMODEL + k0 + c] = t[r * 40 + c];
  }
}

// ---------------------------------------------------------------------------
// Kernel 1: y = x @ W + b (bf16 in, bf16 out), blockIdx.z selects q/k/v.
// 128x128 tile / block, 8 waves of 32x64. Tiles staged by TDM, double
// buffered; inner loop is pure ds_load_b128 + v_wmma.
// z==2 (v) stores transposed vT[D][MTOT] for kernel 2's B-fragments.
// ---------------------------------------------------------------------------
__global__ __launch_bounds__(256) void qkv_gemm_kernel(
    const unsigned short* __restrict__ xb,   // [MTOT][D] bf16
    const unsigned short* __restrict__ Wt,   // [3][D n][D k] bf16
    const float* __restrict__ bq, const float* __restrict__ bk,
    const float* __restrict__ bv,
    unsigned short* __restrict__ qo,
    unsigned short* __restrict__ ko,
    unsigned short* __restrict__ vo) {
  __shared__ __align__(16) unsigned short xs[2][128 * LD1];
  __shared__ __align__(16) unsigned short ws[2][128 * LD1];

  const int z = blockIdx.z;
  const unsigned short* Wz = Wt + (size_t)z * DMODEL * DMODEL;
  const float* bias = (z == 0) ? bq : (z == 1) ? bk : bv;
  const int n0 = blockIdx.x * 128;
  const int m0 = blockIdx.y * 128;
  const int tid  = threadIdx.x;
  const int wave = tid >> 5, lane = tid & 31;
  const int wm = wave >> 1, wn = wave & 1;   // 4x2 wave grid: 32 rows x 64 cols

  v8f acc[2][4] = {};

  if (wave == 0) {
    tdm_load_tile_2d(lds_off(&xs[0][0]), xb + (size_t)m0 * DMODEL,
                     DMODEL, MTOT, DMODEL, 32, 128);
    tdm_load_tile_2d(lds_off(&ws[0][0]), Wz + (size_t)n0 * DMODEL,
                     DMODEL, DMODEL, DMODEL, 32, 128);
  }

  for (int k0 = 0; k0 < DMODEL; k0 += 32) {
    const int buf = (k0 >> 5) & 1;
    if (wave == 0) {
      if (k0 + 32 < DMODEL) {
        tdm_load_tile_2d(lds_off(&xs[buf ^ 1][0]),
                         xb + (size_t)m0 * DMODEL + (k0 + 32),
                         DMODEL, MTOT, DMODEL, 32, 128);
        tdm_load_tile_2d(lds_off(&ws[buf ^ 1][0]),
                         Wz + (size_t)n0 * DMODEL + (k0 + 32),
                         DMODEL, DMODEL, DMODEL, 32, 128);
        __builtin_amdgcn_s_wait_tensorcnt(2);   // current slab landed
      } else {
        __builtin_amdgcn_s_wait_tensorcnt(0);
      }
    }
    __syncthreads();

    v16bf a[2], b[4];
#pragma unroll
    for (int mi = 0; mi < 2; mi++)
      a[mi] = frag_a(&xs[buf][(wm * 32 + mi * 16) * LD1], lane, LD1);
#pragma unroll
    for (int ni = 0; ni < 4; ni++)
      b[ni] = frag_b(&ws[buf][(wn * 64 + ni * 16) * LD1], lane, LD1);
#pragma unroll
    for (int mi = 0; mi < 2; mi++)
#pragma unroll
      for (int ni = 0; ni < 4; ni++)
        acc[mi][ni] = wmma_bf16(a[mi], b[ni], acc[mi][ni]);
    __syncthreads();
  }

  const int cl = lane & 15;
  const int rb = (lane & 16) ? 8 : 0;
#pragma unroll
  for (int mi = 0; mi < 2; mi++) {
#pragma unroll
    for (int ni = 0; ni < 4; ni++) {
      int n  = n0 + wn * 64 + ni * 16 + cl;
      int mb = m0 + wm * 32 + mi * 16 + rb;
      float bsv = bias[n];
      if (z == 2) {
        unsigned short h[8];
#pragma unroll
        for (int r = 0; r < 8; r++) h[r] = f2bf(acc[mi][ni][r] + bsv);
        *(u32x4*)(vo + (size_t)n * MTOT + mb) = *(const u32x4*)h;  // vT[n][m]
      } else {
        unsigned short* o = (z == 0) ? qo : ko;
#pragma unroll
        for (int r = 0; r < 8; r++)
          o[(size_t)(mb + r) * DMODEL + n] = f2bf(acc[mi][ni][r] + bsv);
      }
    }
  }
}

// ---------------------------------------------------------------------------
// Kernel 2: causal flash attention. Block = (batch, 16 query rows).
// 8 waves each own a 128-wide D-slice; q frags persist in registers;
// O[16][128] fp32 in 8 WMMA accumulators per wave. 64-key steps:
// 16 score WMMAs + ds_add_f32 reduction + online softmax (lanes 0-15,
// bank-conflict-free padded rows) + rescale + 16 P@V WMMAs per 3 barriers.
// ---------------------------------------------------------------------------
__global__ __launch_bounds__(256) void attn_kernel(
    const unsigned short* __restrict__ qb,   // [MTOT][D] bf16
    const unsigned short* __restrict__ kb,   // [MTOT][D] bf16
    const unsigned short* __restrict__ vtb,  // [D][MTOT] bf16 (transposed)
    float* __restrict__ out) {               // [B][S][D] fp32
  __shared__ float          Ssc[16 * SLD];   // padded: stride 68 floats
  __shared__ unsigned short Pm[16 * PLD];    // padded: stride 72 bf16
  __shared__ float          mrow[16], lrow[16], arow[16];

  const int qblk  = blockIdx.x;
  const int batch = blockIdx.y;
  const int tid = threadIdx.x, wave = tid >> 5, lane = tid & 31;
  const int cl = lane & 15;
  const int rb = (lane & 16) ? 8 : 0;
  const int qi0 = qblk * 16;
  const size_t mrow0 = (size_t)batch * SEQ + qi0;

  v16bf qa[4];
#pragma unroll
  for (int kt = 0; kt < 4; kt++)
    qa[kt] = frag_a(qb + mrow0 * DMODEL + wave * 128 + kt * 32, lane, DMODEL);

  v8f O[8] = {};

  if (tid < 16) { mrow[tid] = -3.4e38f; lrow[tid] = 0.f; }
  __syncthreads();

  for (int kblk = 0; kblk * 64 <= qi0 + 15; kblk++) {
    for (int j = tid; j < 16 * SLD; j += 256) Ssc[j] = 0.f;
    __syncthreads();

    // Partial scores S[16][64] over this wave's 128-wide D-slice.
    v8f sacc[4] = {};
    const size_t krow0 = (size_t)batch * SEQ + (size_t)kblk * 64;
#pragma unroll
    for (int kt = 0; kt < 4; kt++) {
#pragma unroll
      for (int ni = 0; ni < 4; ni++) {
        v16bf bkf = frag_b(kb + (krow0 + ni * 16) * DMODEL + wave * 128 + kt * 32,
                           lane, DMODEL);
        sacc[ni] = wmma_bf16(qa[kt], bkf, sacc[ni]);
      }
    }
#pragma unroll
    for (int ni = 0; ni < 4; ni++)
#pragma unroll
      for (int r = 0; r < 8; r++)
        atomicAdd(&Ssc[(r + rb) * SLD + ni * 16 + cl], sacc[ni][r]);
    __syncthreads();

    // Online softmax, one lane per query row (padded rows -> no bank clash).
    if (tid < 16) {
      const int row = tid, qi = qi0 + row;
      float mold = mrow[row];
      float sv[64];
#pragma unroll
      for (int j = 0; j < 64; j++) {
        int kj = kblk * 64 + j;
        float s = Ssc[row * SLD + j] * 0.03125f;  // 1/sqrt(1024)
        sv[j] = (kj <= qi) ? s : -3.4e38f;        // causal mask
      }
      float mnew = mold;
#pragma unroll
      for (int j = 0; j < 64; j++) mnew = fmaxf(mnew, sv[j]);
      float al = __expf(mold - mnew);
      float ls = 0.f;
#pragma unroll
      for (int j = 0; j < 64; j++) {
        float p = __expf(sv[j] - mnew);
        ls += p;
        Pm[row * PLD + j] = f2bf(p);
      }
      mrow[row] = mnew;
      lrow[row] = lrow[row] * al + ls;
      arow[row] = al;
    }
    __syncthreads();

    // Rescale running output; accumulate P[16x64] @ V[64x128 slice].
    float av[8];
#pragma unroll
    for (int r = 0; r < 8; r++) av[r] = arow[r + rb];
#pragma unroll
    for (int ni = 0; ni < 8; ni++)
#pragma unroll
      for (int r = 0; r < 8; r++) O[ni][r] *= av[r];

    v16bf pa[2];
#pragma unroll
    for (int kc = 0; kc < 2; kc++)
      pa[kc] = frag_a(&Pm[kc * 32], lane, PLD);
#pragma unroll
    for (int ni = 0; ni < 8; ni++) {
      const unsigned short* vbase =
          vtb + (size_t)(wave * 128 + ni * 16) * MTOT +
          (size_t)batch * SEQ + (size_t)kblk * 64;
#pragma unroll
      for (int kc = 0; kc < 2; kc++)
        O[ni] = wmma_bf16(pa[kc], frag_b(vbase + kc * 32, lane, MTOT), O[ni]);
    }
  }

  float linv[8];
#pragma unroll
  for (int r = 0; r < 8; r++) linv[r] = 1.0f / lrow[r + rb];
#pragma unroll
  for (int ni = 0; ni < 8; ni++) {
    int col = wave * 128 + ni * 16 + cl;
#pragma unroll
    for (int r = 0; r < 8; r++) {
      int row = qi0 + r + rb;
      out[((size_t)batch * SEQ + row) * DMODEL + col] = O[ni][r] * linv[r];
    }
  }
}

// ---------------------------------------------------------------------------
extern "C" void kernel_launch(void* const* d_in, const int* in_sizes, int n_in,
                              void* d_out, int out_size, void* d_ws, size_t ws_size,
                              hipStream_t stream) {
  (void)in_sizes; (void)n_in; (void)out_size; (void)ws_size;
  const float* x  = (const float*)d_in[0];
  const float* Wq = (const float*)d_in[1];
  const float* bq = (const float*)d_in[2];
  const float* Wk = (const float*)d_in[3];
  const float* bk = (const float*)d_in[4];
  const float* Wv = (const float*)d_in[5];
  const float* bv = (const float*)d_in[6];

  // Workspace layout (bf16): q | k | vT | xb | Wt   = 16+16+16+16+6 MB
  unsigned short* qo = (unsigned short*)d_ws;
  unsigned short* ko = qo + (size_t)MTOT * DMODEL;
  unsigned short* vo = ko + (size_t)MTOT * DMODEL;
  unsigned short* xb = vo + (size_t)MTOT * DMODEL;
  unsigned short* wt = xb + (size_t)MTOT * DMODEL;

  convert_x_kernel<<<(MTOT * DMODEL) / (256 * 8), 256, 0, stream>>>(x, xb);
  convert_w_kernel<<<dim3(DMODEL / 32, DMODEL / 32, 3), 256, 0, stream>>>(Wq, Wk, Wv, wt);

  dim3 g1(DMODEL / 128, MTOT / 128, 3);  // (8, 64, 3)
  qkv_gemm_kernel<<<g1, 256, 0, stream>>>(xb, wt, bq, bk, bv, qo, ko, vo);

  dim3 g2(SEQ / 16, NBATCH);             // (128, 4)
  attn_kernel<<<g2, 256, 0, stream>>>(qo, ko, vo, (float*)d_out);
}
